// MultiHeadAttentionLayer_4501125726695
// MI455X (gfx1250) — matrix operand order repaired and checked by reading, not compiled, not run on hardware
//
#include <hip/hip_runtime.h>

typedef __attribute__((ext_vector_type(16))) __bf16 v16bf;
typedef __attribute__((ext_vector_type(8)))  __bf16 v8bf;
typedef __attribute__((ext_vector_type(8)))  float  v8f;

static __device__ inline v8f wmma_bf16(v16bf a, v16bf b, v8f c) {
  // D = A(16x32 bf16) x B(32x16 bf16) + C(16x16 f32)
  return __builtin_amdgcn_wmma_f32_16x16x32_bf16(false, a, false, b, (short)0, c, false, false);
}
static __device__ inline v16bf cat8(v8bf lo, v8bf hi) {
  return __builtin_shufflevector(lo, hi, 0,1,2,3,4,5,6,7,8,9,10,11,12,13,14,15);
}
static __device__ inline v8f vzero8() {
  v8f z = {0.f,0.f,0.f,0.f,0.f,0.f,0.f,0.f};
  return z;
}
static __device__ inline void wait_async0() {
#if __has_builtin(__builtin_amdgcn_s_wait_asynccnt)
  __builtin_amdgcn_s_wait_asynccnt(0);
#else
  asm volatile("s_wait_asynccnt 0x0" ::: "memory");
#endif
}

// ---------------------------------------------------------------------------
// Sinusoid table: TAB[p][h], p in [0,255) used, row 255 zero. bf16.
// table[p, h] = h even ? sin(p / 10000^(2*(h/2)/64)) : cos(...)
// ---------------------------------------------------------------------------
__global__ void table_kernel(__bf16* __restrict__ TAB) {
  int p = blockIdx.x;        // 0..255
  int h = threadIdx.x;       // 0..63
  float val = 0.f;
  if (p < 255) {
    float denom = __powf(10000.0f, (float)(2 * (h >> 1)) / 64.0f);
    float ang = (float)p / denom;
    val = (h & 1) ? __cosf(ang) : __sinf(ang);
  }
  TAB[p * 64 + h] = (__bf16)val;
}

// ---------------------------------------------------------------------------
// f32 -> bf16 elementwise convert
// ---------------------------------------------------------------------------
__global__ void cvt_kernel(const float* __restrict__ src, __bf16* __restrict__ dst, int n) {
  int i = blockIdx.x * 256 + threadIdx.x;
  if (i < n) dst[i] = (__bf16)src[i];
}

// ---------------------------------------------------------------------------
// Tiled GEMM: C = A[M,K] @ B[K,N] + bias[N]
//   A, B bf16 row-major. Block = 128 threads (4 waves), 64x64 tile, BK=32.
//   A tile staged via GLOBAL_LOAD_ASYNC_TO_LDS_B128 (ASYNCcnt path).
//   mode 0: store bf16 into [B,NH,S,HD] head-split layout (QKV)
//   mode 1: store f32 row-major (output projection)
// ---------------------------------------------------------------------------
__global__ void gemm_kernel(const __bf16* __restrict__ A, const __bf16* __restrict__ B,
                            const float* __restrict__ bias, void* __restrict__ Cout,
                            int N, int K, int mode) {
  __shared__ __align__(16) __bf16 at[64][32];   // A tile, row-major
  __shared__ __align__(16) __bf16 btT[64][32];  // B tile transposed: [n][k]
  const int wave = threadIdx.x >> 5;
  const int lane = threadIdx.x & 31;
  const int hl = lane >> 4;       // half of wave
  const int c  = lane & 15;       // lane within half
  const int m0 = blockIdx.x * 64;
  const int n0 = blockIdx.y * 64;

  v8f acc[4];
#pragma unroll
  for (int nc = 0; nc < 4; ++nc) acc[nc] = vzero8();

  for (int kk = 0; kk < K; kk += 32) {
    __syncthreads();
    {  // stage A: 64x32, 32B/thread, async DMA global -> LDS (no VGPR round-trip)
      int t = threadIdx.x;
      int r = t >> 1, cc = (t & 1) * 16;
      const __bf16* src = A + (size_t)(m0 + r) * K + kk + cc;
      unsigned lds_off = (unsigned)(size_t)&at[r][cc];
      // IOFFSET applies to both LDS and memory address (ISA 10.4.4), so the
      // second b128 covers bytes [16,32) on both sides.
      asm volatile(
          "global_load_async_to_lds_b128 %0, %1, off\n\t"
          "global_load_async_to_lds_b128 %0, %1, off offset:16"
          :: "v"(lds_off), "v"((const void*)src) : "memory");
      if (kk + 32 < K) __builtin_prefetch(src + 32, 0, 1);  // global_prefetch_b8
    }
    {  // stage B transposed: 32x64 -> btT[n][k]
      int t = threadIdx.x;
      int kr = t >> 2, ncs = (t & 3) * 16;
      const __bf16* src = B + (size_t)(kk + kr) * N + n0 + ncs;
#pragma unroll
      for (int u = 0; u < 16; ++u) btT[ncs + u][kr] = src[u];
      if (kk + 32 < K) __builtin_prefetch(src + 32 * (size_t)N, 0, 1);
    }
    wait_async0();          // ASYNCcnt -> 0 before the barrier
    __syncthreads();
    // A fragment: row = 16*wave + (lane&15); k = e + 16*hl
    v16bf af = cat8(*(const v8bf*)&at[16 * wave + c][16 * hl],
                    *(const v8bf*)&at[16 * wave + c][16 * hl + 8]);
#pragma unroll
    for (int nc = 0; nc < 4; ++nc) {
      // B fragment: n = lane&15, k = e + 16*hl  -> contiguous in btT
      v16bf bf = *(const v16bf*)&btT[nc * 16 + c][16 * hl];
      acc[nc] = wmma_bf16(af, bf, acc[nc]);
    }
  }
  __syncthreads();

#pragma unroll
  for (int nc = 0; nc < 4; ++nc) {
#pragma unroll
    for (int g = 0; g < 8; ++g) {
      int r  = m0 + 16 * wave + g + 8 * hl;   // global row
      int cg = n0 + nc * 16 + c;              // global col
      float v = acc[nc][g] + bias[cg];
      if (mode == 0) {
        int b = r >> 10, i = r & 1023;
        int head = cg >> 6, hd = cg & 63;
        ((__bf16*)Cout)[(size_t)((b * 16 + head) * 1024 + i) * 64 + hd] = (__bf16)v;
      } else {
        ((float*)Cout)[(size_t)r * N + cg] = v;
      }
    }
  }
}

// ---------------------------------------------------------------------------
// QR[row, p] = sum_h Q[row, h] * TAB[p, h]   (q @ table^T), f32 out.
// M = 65536 rows, N = 256, K = 64. Block = 128 threads (4 waves), no LDS.
// ---------------------------------------------------------------------------
__global__ void qr_kernel(const __bf16* __restrict__ Q, const __bf16* __restrict__ TAB,
                          float* __restrict__ QR) {
  const int wave = threadIdx.x >> 5;
  const int lane = threadIdx.x & 31;
  const int hl = lane >> 4, c = lane & 15;
  const int i0 = blockIdx.x * 64 + wave * 16;
  const int n0 = blockIdx.y * 64;
  const __bf16* qrow = Q + (size_t)(i0 + c) * 64;

  v8f acc[4];
#pragma unroll
  for (int nc = 0; nc < 4; ++nc) acc[nc] = vzero8();

#pragma unroll
  for (int kk = 0; kk < 2; ++kk) {
    v16bf aq = cat8(*(const v8bf*)(qrow + 32 * kk + 16 * hl),
                    *(const v8bf*)(qrow + 32 * kk + 16 * hl + 8));
#pragma unroll
    for (int nc = 0; nc < 4; ++nc) {
      // B[k][n] = TAB[n0+n, k]: contiguous 16 bf16 from table row
      v16bf bt = *(const v16bf*)(TAB + (size_t)(n0 + nc * 16 + c) * 64 + 32 * kk + 16 * hl);
      acc[nc] = wmma_bf16(aq, bt, acc[nc]);
    }
  }
#pragma unroll
  for (int nc = 0; nc < 4; ++nc)
#pragma unroll
    for (int g = 0; g < 8; ++g)
      QR[(size_t)(i0 + g + 8 * hl) * 256 + n0 + nc * 16 + c] = acc[nc][g];
}

// ---------------------------------------------------------------------------
// Attention: per (b, head, 16-row tile). Block = 64 threads (2 waves).
// Two-pass softmax; rel-key via QR gather, rel-value via PS bucket + WMMA.
// ---------------------------------------------------------------------------
__global__ void attn_kernel(const __bf16* __restrict__ Q, const __bf16* __restrict__ K,
                            const __bf16* __restrict__ V, const float* __restrict__ QR,
                            const __bf16* __restrict__ TAB, const float* __restrict__ mask,
                            __bf16* __restrict__ CTX) {
  __shared__ __align__(16) float  ps[2][16][256];    // prob bucket sums, per wave (32KB)
  __shared__ __align__(16) __bf16 vtile[64][32];     // V/TAB tile transposed [n][k] (4KB)
  __shared__ __align__(16) __bf16 ptile[2][16][32];  // prob tile per wave (2KB)

  const int wave = threadIdx.x >> 5;
  const int lane = threadIdx.x & 31;
  const int hl = lane >> 4, c = lane & 15;
  const int bh = blockIdx.x >> 5;                  // 0..63
  const int b = bh >> 4;
  const int i0 = ((blockIdx.x & 31) << 5) + (wave << 4);
  const __bf16* Qb = Q + (size_t)bh * 1024 * 64;
  const __bf16* Kb = K + (size_t)bh * 1024 * 64;
  const __bf16* Vb = V + (size_t)bh * 1024 * 64;
  const float* QRb = QR + (size_t)bh * 1024 * 256;
  const float* mb  = mask + (size_t)b * 1024;

  // zero bucket accumulator
  for (int idx = lane; idx < 16 * 256; idx += 32)
    ((float*)ps[wave])[idx] = 0.f;

  // Q fragments (row = i0 + lane%16), kept resident for whole kernel
  v16bf aq[2];
  {
    const __bf16* qrow = Qb + (size_t)(i0 + c) * 64;
#pragma unroll
    for (int kk = 0; kk < 2; ++kk)
      aq[kk] = cat8(*(const v8bf*)(qrow + 32 * kk + 16 * hl),
                    *(const v8bf*)(qrow + 32 * kk + 16 * hl + 8));
  }

  // ---- Pass A: row max + exp-sum (online, scalar stats only) ----
  float rm[8], rl[8];
#pragma unroll
  for (int g = 0; g < 8; ++g) { rm[g] = -1e30f; rl[g] = 0.f; }

  for (int jt = 0; jt < 64; ++jt) {
    v8f sacc = vzero8();
#pragma unroll
    for (int kk = 0; kk < 2; ++kk) {
      v16bf bk = *(const v16bf*)(Kb + (size_t)(16 * jt + c) * 64 + 32 * kk + 16 * hl);
      sacc = wmma_bf16(aq[kk], bk, sacc);
    }
#pragma unroll
    for (int g = 0; g < 8; ++g) {
      int i = i0 + g + 8 * hl;
      int j = 16 * jt + c;
      int d = j - i; d = d > 127 ? 127 : (d < -127 ? -127 : d);
      float s = sacc[g] + QRb[(size_t)i * 256 + (d + 127)];
      s *= 0.125f;                                   // 1/sqrt(64)
      s += (1.0f - mb[j]) * -10000.0f;
      // tile max over the 16 lanes of this half
      float tm = s;
      tm = fmaxf(tm, __shfl_xor(tm, 1, 16));
      tm = fmaxf(tm, __shfl_xor(tm, 2, 16));
      tm = fmaxf(tm, __shfl_xor(tm, 4, 16));
      tm = fmaxf(tm, __shfl_xor(tm, 8, 16));
      float mnew = fmaxf(rm[g], tm);
      float pe = __expf(s - mnew);
      pe += __shfl_xor(pe, 1, 16);
      pe += __shfl_xor(pe, 2, 16);
      pe += __shfl_xor(pe, 4, 16);
      pe += __shfl_xor(pe, 8, 16);
      rl[g] = rl[g] * __expf(rm[g] - mnew) + pe;
      rm[g] = mnew;
    }
  }
  float rinv[8];
#pragma unroll
  for (int g = 0; g < 8; ++g) rinv[g] = 1.0f / rl[g];

  // ---- Pass B: normalized probs -> P@V (WMMA) + PS buckets ----
  v8f oacc[4];
#pragma unroll
  for (int nc = 0; nc < 4; ++nc) oacc[nc] = vzero8();

  for (int jp = 0; jp < 32; ++jp) {
    __syncthreads();
    {  // stage V rows [32jp, 32jp+32) transposed into vtile[n][k]
      int t = threadIdx.x;
      int kr = t >> 1, nc0 = (t & 1) * 32;
      const __bf16* vrow = Vb + (size_t)(32 * jp + kr) * 64 + nc0;
#pragma unroll
      for (int u = 0; u < 32; ++u) vtile[nc0 + u][kr] = vrow[u];
    }
#pragma unroll
    for (int t2 = 0; t2 < 2; ++t2) {
      int jt = 2 * jp + t2;
      v8f sacc = vzero8();
#pragma unroll
      for (int kk = 0; kk < 2; ++kk) {
        v16bf bk = *(const v16bf*)(Kb + (size_t)(16 * jt + c) * 64 + 32 * kk + 16 * hl);
        sacc = wmma_bf16(aq[kk], bk, sacc);
      }
#pragma unroll
      for (int g = 0; g < 8; ++g) {
        int i = i0 + g + 8 * hl;
        int j = 16 * jt + c;
        int d = j - i; d = d > 127 ? 127 : (d < -127 ? -127 : d);
        float s = sacc[g] + QRb[(size_t)i * 256 + (d + 127)];
        s *= 0.125f;
        s += (1.0f - mb[j]) * -10000.0f;
        float p = __expf(s - rm[g]) * rinv[g];       // final normalized prob
        atomicAdd(&ps[wave][g + 8 * hl][d + 127], p);  // ds_add_f32
        ptile[wave][g + 8 * hl][c + 16 * t2] = (__bf16)p;
      }
    }
    __syncthreads();
    // P fragment (16x32) from ptile; V fragments from vtile
    v16bf ap = cat8(*(const v8bf*)&ptile[wave][c][16 * hl],
                    *(const v8bf*)&ptile[wave][c][16 * hl + 8]);
#pragma unroll
    for (int nc = 0; nc < 4; ++nc) {
      v16bf bv = *(const v16bf*)&vtile[nc * 16 + c][16 * hl];
      oacc[nc] = wmma_bf16(ap, bv, oacc[nc]);
    }
  }

  // ---- PS @ TAB: relative-value context ----
  for (int kk = 0; kk < 8; ++kk) {
    __syncthreads();
    {  // stage TAB rows [32kk, 32kk+32) transposed into vtile
      int t = threadIdx.x;
      int kr = t >> 1, nc0 = (t & 1) * 32;
      const __bf16* trow = TAB + (size_t)(32 * kk + kr) * 64 + nc0;
#pragma unroll
      for (int u = 0; u < 32; ++u) vtile[nc0 + u][kr] = trow[u];
    }
    __syncthreads();
    v16bf ap;
#pragma unroll
    for (int e = 0; e < 16; ++e) {
      int kloc = e + 16 * hl;
      ap[e] = (__bf16)ps[wave][c][32 * kk + kloc];
    }
#pragma unroll
    for (int nc = 0; nc < 4; ++nc) {
      v16bf bt = *(const v16bf*)&vtile[nc * 16 + c][16 * hl];
      oacc[nc] = wmma_bf16(ap, bt, oacc[nc]);
    }
  }

  // ---- store ctx as bf16 in [B, S, H] with H = head*64 + hd ----
  const int head = bh & 15;
#pragma unroll
  for (int nc = 0; nc < 4; ++nc)
#pragma unroll
    for (int g = 0; g < 8; ++g) {
      int i = i0 + g + 8 * hl;
      int col = head * 64 + nc * 16 + c;
      CTX[(size_t)(b * 1024 + i) * 1024 + col] = (__bf16)oacc[nc][g];
    }
}

// ---------------------------------------------------------------------------
// Launch
// ---------------------------------------------------------------------------
extern "C" void kernel_launch(void* const* d_in, const int* in_sizes, int n_in,
                              void* d_out, int out_size, void* d_ws, size_t ws_size,
                              hipStream_t stream) {
  const float* hs   = (const float*)d_in[0];   // [4,1024,1024]
  const float* mask = (const float*)d_in[1];   // [4,1,1,1024]
  const float* Wq   = (const float*)d_in[2];
  const float* bq   = (const float*)d_in[3];
  const float* Wk   = (const float*)d_in[4];
  const float* bk   = (const float*)d_in[5];
  const float* Wv   = (const float*)d_in[6];
  const float* bv   = (const float*)d_in[7];
  const float* Wo   = (const float*)d_in[8];
  const float* bo   = (const float*)d_in[9];

  char* ws = (char*)d_ws;
  auto alloc = [&](size_t bytes) -> char* {
    char* p = ws;
    ws += (bytes + 255) & ~(size_t)255;
    return p;
  };
  __bf16* Xbf   = (__bf16*)alloc((size_t)4096 * 1024 * 2);
  __bf16* Wqbf  = (__bf16*)alloc((size_t)1024 * 1024 * 2);
  __bf16* Wkbf  = (__bf16*)alloc((size_t)1024 * 1024 * 2);
  __bf16* Wvbf  = (__bf16*)alloc((size_t)1024 * 1024 * 2);
  __bf16* Wobf  = (__bf16*)alloc((size_t)1024 * 1024 * 2);
  __bf16* Qbf   = (__bf16*)alloc((size_t)65536 * 64 * 2);
  __bf16* Kbf   = (__bf16*)alloc((size_t)65536 * 64 * 2);
  __bf16* Vbf   = (__bf16*)alloc((size_t)65536 * 64 * 2);
  __bf16* TAB   = (__bf16*)alloc((size_t)256 * 64 * 2);
  float*  QRf   = (float*) alloc((size_t)65536 * 256 * 4);
  __bf16* CTXbf = (__bf16*)alloc((size_t)4096 * 1024 * 2);

  // sinusoid table
  table_kernel<<<dim3(256), dim3(64), 0, stream>>>(TAB);
  // bf16 conversions
  cvt_kernel<<<dim3(16384), dim3(256), 0, stream>>>(hs, Xbf, 4194304);
  cvt_kernel<<<dim3(4096),  dim3(256), 0, stream>>>(Wq, Wqbf, 1048576);
  cvt_kernel<<<dim3(4096),  dim3(256), 0, stream>>>(Wk, Wkbf, 1048576);
  cvt_kernel<<<dim3(4096),  dim3(256), 0, stream>>>(Wv, Wvbf, 1048576);
  cvt_kernel<<<dim3(4096),  dim3(256), 0, stream>>>(Wo, Wobf, 1048576);
  // QKV projections (head-split bf16 outputs)
  gemm_kernel<<<dim3(64, 16), dim3(128), 0, stream>>>(Xbf, Wqbf, bq, (void*)Qbf, 1024, 1024, 0);
  gemm_kernel<<<dim3(64, 16), dim3(128), 0, stream>>>(Xbf, Wkbf, bk, (void*)Kbf, 1024, 1024, 0);
  gemm_kernel<<<dim3(64, 16), dim3(128), 0, stream>>>(Xbf, Wvbf, bv, (void*)Vbf, 1024, 1024, 0);
  // QR = q @ table^T
  qr_kernel<<<dim3(1024, 4), dim3(128), 0, stream>>>(Qbf, TAB, QRf);
  // attention
  attn_kernel<<<dim3(2048), dim3(64), 0, stream>>>(Qbf, Kbf, Vbf, QRf, TAB, mask, CTXbf);
  // output projection -> f32
  gemm_kernel<<<dim3(64, 16), dim3(128), 0, stream>>>(CTXbf, Wobf, bo, d_out, 1024, 1024, 1);

  (void)in_sizes; (void)n_in; (void)out_size; (void)ws_size;
}
